// SkipLSTMLayer_57217554317373
// MI455X (gfx1250) — matrix operand order repaired and checked by reading, not compiled, not run on hardware
//
#include <hip/hip_runtime.h>
#include <hip/hip_bf16.h>

typedef __attribute__((ext_vector_type(16))) __bf16 v16bf;
typedef __attribute__((ext_vector_type(8)))  float  v8f;

#define B_  256
#define T_  2048
#define D_  128
#define H_  128
#define G4  512      // 4*H
#define KC  256      // D + H (fused GEMM K)
#define LDA 272      // padded LDS A row stride (bf16 elems): 16B-aligned rows, low bank conflict

#if __has_builtin(__builtin_amdgcn_tanhf)
__device__ __forceinline__ float fast_tanh(float x)    { return __builtin_amdgcn_tanhf(x); }
__device__ __forceinline__ float fast_sigmoid(float x) { return fmaf(0.5f, __builtin_amdgcn_tanhf(0.5f * x), 0.5f); }
#elif __has_builtin(__builtin_amdgcn_tanh_f32)
__device__ __forceinline__ float fast_tanh(float x)    { return __builtin_amdgcn_tanh_f32(x); }
__device__ __forceinline__ float fast_sigmoid(float x) { return fmaf(0.5f, __builtin_amdgcn_tanh_f32(0.5f * x), 0.5f); }
#else
__device__ __forceinline__ float fast_sigmoid(float x) { return 1.0f / (1.0f + __expf(-x)); }
__device__ __forceinline__ float fast_tanh(float x)    { return 1.0f - 2.0f / (1.0f + __expf(2.0f * x)); }
#endif

// Phase 0: build W_T[n][k] (bf16), n in [0,512), k in [0,256).
// k < 128 -> kernel[k][n] ; k >= 128 -> rec_kernel[k-128][n]
__global__ void build_weights_kernel(const float* __restrict__ kern,
                                     const float* __restrict__ rkern,
                                     __bf16* __restrict__ wt) {
    int idx = blockIdx.x * blockDim.x + threadIdx.x;   // 512*256 total
    int n = idx >> 8;
    int k = idx & 255;
    float v = (k < D_) ? kern[(size_t)k * G4 + n] : rkern[(size_t)(k - D_) * G4 + n];
    wt[(size_t)n * KC + k] = (__bf16)v;
}

// One workgroup = 16 batch rows, 8 waves; wave w owns h-columns [16w,16w+16) of
// all four gates. Weights resident in VGPRs; ONE barrier per time step.
__global__ __launch_bounds__(256, 1)
void skiplstm_kernel(const float* __restrict__ x,
                     const float* __restrict__ bias,
                     const float* __restrict__ dw,
                     const float* __restrict__ db,
                     const __bf16* __restrict__ wt,
                     float* __restrict__ out) {
    __shared__ __bf16 ldsA[2][16 * LDA];   // double-buffered fused A operand [x_t | h_b]
    __shared__ float  ldsDot[3][16];       // triple-buffered delta dot accumulators

    const int tid  = threadIdx.x;
    const int wave = tid >> 5;
    const int lane = tid & 31;
    const int lh   = lane & 15;         // column within 16x16 tile / A row
    const int hi   = lane >> 4;         // half-wave selector
    const int m0   = blockIdx.x * 16;   // batch row base
    const int hw   = wave * 16;         // this wave's h-column base

    const int ncol[4] = { hw, 128 + hw, 256 + hw, 384 + hw };  // i, f, g, o

    // ---- resident B fragments: 8 k-steps x 4 n-tiles (256 VGPRs) ----
    // B operand, this code's contiguous k order (same as A): lane N = lh, K base = hi*16, VGPR v = {K+2v, K+2v+1}
    v16bf bw[8][4];
#pragma unroll
    for (int k = 0; k < 8; ++k)
#pragma unroll
        for (int n = 0; n < 4; ++n) {
            const __bf16* p = wt + (size_t)(ncol[n] + lh) * KC + k * 32 + hi * 16;
            bw[k][n] = *(const v16bf*)p;
        }

    float bcol[4];
#pragma unroll
    for (int n = 0; n < 4; ++n) bcol[n] = bias[ncol[n] + lh];
    const float dwh = dw[hw + lh];          // dense_w, h_b part
    const float dwc = dw[128 + hw + lh];    // dense_w, c_b part
    const float dbv = db[0];

    // per-wave carried state (C-layout: row = hi*8 + r, col = hw + lh)
    float hb[8], cb[8], uu[8];
#pragma unroll
    for (int r = 0; r < 8; ++r) { hb[r] = 0.f; cb[r] = 0.f; uu[r] = 1.0f; }

    // x staging: thread loads batch row xi, dims [xd0, xd0+8)
    const int xi  = tid >> 4;
    const int xd0 = (tid & 15) * 8;
    const float* xrow = x + ((size_t)(m0 + xi) * T_) * D_ + xd0;

    float* orow[8];
#pragma unroll
    for (int r = 0; r < 8; ++r)
        orow[r] = out + ((size_t)(m0 + hi * 8 + r) * T_) * H_ + hw + lh;

    // ---- prologue: stage A buffer 0 for t=0, clear all dot buffers ----
    {
        const float4 a = *(const float4*)(xrow);
        const float4 b = *(const float4*)(xrow + 4);
        union { __bf16 h[8]; uint4 u; } px;
        px.h[0] = (__bf16)a.x; px.h[1] = (__bf16)a.y;
        px.h[2] = (__bf16)a.z; px.h[3] = (__bf16)a.w;
        px.h[4] = (__bf16)b.x; px.h[5] = (__bf16)b.y;
        px.h[6] = (__bf16)b.z; px.h[7] = (__bf16)b.w;
        *(uint4*)&ldsA[0][xi * LDA + xd0] = px.u;
#pragma unroll
        for (int r = 0; r < 8; ++r)
            ldsA[0][(hi * 8 + r) * LDA + 128 + hw + lh] = (__bf16)0.f;
    }
    if (tid < 48) ldsDot[tid >> 4][tid & 15] = 0.f;
    __syncthreads();

    for (int t = 0; t < T_; ++t) {
        // ---- issue x_{t+1} global loads early (fully hidden behind GEMM) ----
        const int tn = (t + 1 < T_) ? (t + 1) : t;
        const float4 xa = *(const float4*)(xrow + (size_t)tn * D_);
        const float4 xb = *(const float4*)(xrow + (size_t)tn * D_ + 4);

        // ---- skip-gate update from step t-1's dot; ug for step t (per lane) ----
        float ug[8];
        if (t > 0) {
            const float* dotp = &ldsDot[(t + 2) % 3][hi * 8];   // (t-1) mod 3
            const float4 d0 = *(const float4*)(dotp);
            const float4 d1 = *(const float4*)(dotp + 4);
            const float dotv[8] = { d0.x, d0.y, d0.z, d0.w, d1.x, d1.y, d1.z, d1.w };
#pragma unroll
            for (int r = 0; r < 8; ++r) {
                const float ugp   = rintf(uu[r]);                 // ug used in step t-1
                const float delta = fast_sigmoid(dotv[r] + dbv);
                uu[r] = ugp * delta + (1.f - ugp) * (uu[r] + fminf(delta, 1.f - uu[r]));
                ug[r] = rintf(uu[r]);                             // ug for step t
            }
        } else {
#pragma unroll
            for (int r = 0; r < 8; ++r) ug[r] = 1.0f;             // round(u0=1)
        }

        // ---- z = [x_t, h] @ W + bias : 8 k-steps x 4 n-tiles of WMMA ----
        const __bf16* Abuf = ldsA[t & 1];
        v8f acc[4];
#pragma unroll
        for (int n = 0; n < 4; ++n)
#pragma unroll
            for (int r = 0; r < 8; ++r) acc[n][r] = bcol[n];

#pragma unroll
        for (int k = 0; k < 8; ++k) {
            // A layout (16-bit 16x32): row = lh, K = k*32 + hi*16 + s (contiguous, matches B)
            union { uint4 u[2]; v16bf v; } af;
            const char* base = (const char*)Abuf + lh * (LDA * 2) + k * 64 + hi * 32;
            af.u[0] = *(const uint4*)(base);
            af.u[1] = *(const uint4*)(base + 16);
#pragma unroll
            for (int n = 0; n < 4; ++n)
                acc[n] = __builtin_amdgcn_wmma_f32_16x16x32_bf16(
                    false, af.v, false, bw[k][n], (short)0, acc[n], false, false);
        }

        // ---- gates, cell/hidden update, skip gating ----
        float part[8];
#pragma unroll
        for (int r = 0; r < 8; ++r) {
            const float ig = fast_sigmoid(acc[0][r]);
            const float fg = fast_sigmoid(acc[1][r]);
            const float gg = fast_tanh(acc[2][r]);
            const float og = fast_sigmoid(acc[3][r]);
            const float cn = fg * cb[r] + ig * gg;
            const float hn = og * fast_tanh(cn);
            orow[r][(size_t)t * H_] = hn;                    // output is h_new
            const float u = ug[r];
            hb[r] = u * hb[r] + (1.f - u) * hn;              // h_b
            cb[r] = u * cb[r] + (1.f - u) * cn;              // c_b
            part[r] = hb[r] * dwh + cb[r] * dwc;             // delta partial
        }

        // ---- stage next A buffer (x_{t+1} + new h_b) ----
        __bf16* Anext = ldsA[(t + 1) & 1];
        {
            union { __bf16 h[8]; uint4 u; } px;
            px.h[0] = (__bf16)xa.x; px.h[1] = (__bf16)xa.y;
            px.h[2] = (__bf16)xa.z; px.h[3] = (__bf16)xa.w;
            px.h[4] = (__bf16)xb.x; px.h[5] = (__bf16)xb.y;
            px.h[6] = (__bf16)xb.z; px.h[7] = (__bf16)xb.w;
            *(uint4*)&Anext[xi * LDA + xd0] = px.u;
        }
#pragma unroll
        for (int r = 0; r < 8; ++r)
            Anext[(hi * 8 + r) * LDA + 128 + hw + lh] = (__bf16)hb[r];

        // ---- reduce delta partials over this wave's 16 columns ----
#pragma unroll
        for (int m = 1; m < 16; m <<= 1)
#pragma unroll
            for (int r = 0; r < 8; ++r)
                part[r] += __shfl_xor(part[r], m, 16);

        if (lh == 0) {
            float* dst = &ldsDot[t % 3][hi * 8];
#pragma unroll
            for (int r = 0; r < 8; ++r) atomicAdd(&dst[r], part[r]);
        }
        if (tid < 16) ldsDot[(t + 1) % 3][tid] = 0.f;        // clear for step t+1

        __syncthreads();   // single barrier per time step
    }
}

extern "C" void kernel_launch(void* const* d_in, const int* in_sizes, int n_in,
                              void* d_out, int out_size, void* d_ws, size_t ws_size,
                              hipStream_t stream) {
    const float* x     = (const float*)d_in[0];   // (B, T, D)
    const float* kern  = (const float*)d_in[1];   // (D, 4H)
    const float* rkern = (const float*)d_in[2];   // (H, 4H)
    const float* bias  = (const float*)d_in[3];   // (4H,)
    const float* dw    = (const float*)d_in[4];   // (2H, 1)
    const float* db    = (const float*)d_in[5];   // (1,)
    __bf16* wt = (__bf16*)d_ws;                   // W_T: 512 x 256 bf16 = 256 KB
    float* out = (float*)d_out;                   // (B, T, H)

    build_weights_kernel<<<(G4 * KC) / 256, 256, 0, stream>>>(kern, rkern, wt);
    skiplstm_kernel<<<B_ / 16, 256, 0, stream>>>(x, bias, dw, db, wt, out);
}
